// SoftMOE_56831007261006
// MI455X (gfx1250) — matrix operand order repaired and checked by reading, not compiled, not run on hardware
//
#include <hip/hip_runtime.h>
#include <hip/hip_bf16.h>

typedef __attribute__((ext_vector_type(16))) __bf16 bf16x16;
typedef __attribute__((ext_vector_type(8)))  __bf16 bf16x8;
typedef __attribute__((ext_vector_type(8)))  float  v8f;
typedef __attribute__((ext_vector_type(4)))  float  f32x4;

static constexpr int BB = 4, SS = 8192, DD = 768, EE = 64;

// Per-lane fragment region: 16 bf16 used, padded to 24 (48B) to keep 16B
// alignment for ds_load_b128 while limiting bank conflicts to the 2-phase min.
static constexpr int FS   = 24;        // frag stride (elems) per lane
static constexpr int SUBT = 32 * FS;   // elems per 16x32(A) / 32x16(B) subtile

__device__ __forceinline__ __bf16 f2bf(float f) { return (__bf16)f; }

// A subtile (16x32, MxK): lanes 0-15 hold K {0..7,16..23}, lanes 16-31 hold
// K {8..15,24..31}; element index within lane = (k&7) + 8*(k>>4).
__device__ __forceinline__ int a_off(int m, int k) {
  return (m + (((k >> 3) & 1) << 4)) * FS + ((k & 7) + ((k >> 4) << 3));
}
// B subtile (32x16, KxN): lanes 0-15 hold K {0..7,16..23}, lanes 16-31 hold
// K {8..15,24..31}; element index within lane = (k&7) + 8*(k>>4).
__device__ __forceinline__ int b_off(int k, int n) {
  return (n + (((k >> 3) & 1) << 4)) * FS + ((k & 7) + ((k >> 4) << 3));
}

// Fragment load: two ds_load_b128 per lane (contiguous 32B region).
__device__ __forceinline__ bf16x16 load_frag(const __bf16* S, int lane) {
  const bf16x8* p = (const bf16x8*)(S + lane * FS);
  bf16x8 lo = p[0];
  bf16x8 hi = p[1];
  return __builtin_shufflevector(lo, hi, 0, 1, 2, 3, 4, 5, 6, 7,
                                 8, 9, 10, 11, 12, 13, 14, 15);
}

__device__ __forceinline__ v8f wmma_bf16(bf16x16 a, bf16x16 b, v8f c) {
  return __builtin_amdgcn_wmma_f32_16x16x32_bf16(false, a, false, b, (short)0, c, false, false);
}

__device__ __forceinline__ f32x4 ldg4(const float* p) { return *(const f32x4*)p; }
__device__ __forceinline__ f32x4 ldg4_nt(const float* p) {
  return __builtin_nontemporal_load((const f32x4*)p);
}
__device__ __forceinline__ void st4(__bf16* d, f32x4 v) {
  d[0] = f2bf(v.x); d[1] = f2bf(v.y); d[2] = f2bf(v.z); d[3] = f2bf(v.w);
}

// ---------------- K1: gating = x @ Wg  [BS x E] ----------------
// 256 thr (8 waves), tile 32(M) x 64(N), K=768, double-buffered
__global__ __launch_bounds__(256) void softmoe_gating(const float* __restrict__ x,
                                                      const float* __restrict__ Wg,
                                                      float* __restrict__ gating) {
  __shared__ __align__(16) __bf16 As[2][2 * SUBT];
  __shared__ __align__(16) __bf16 Bs[2][4 * SUBT];
  int tid = threadIdx.x, lane = tid & 31, wave = tid >> 5;
  int mi = wave >> 2, ni = wave & 3;
  int m0 = blockIdx.x * 32;

  // A fill: one float4 per thread, k-contiguous -> contiguous frag elems
  int ar = tid >> 3, ac = (tid & 7) * 4;
  const float* agp = &x[(size_t)(m0 + ar) * DD + ac];
  int aoff = (ar >> 4) * SUBT + a_off(ar & 15, ac);
  // B fill: two float4 per thread (rows br, br+16)
  int br = tid >> 4, bc = (tid & 15) * 4;
  const float* bgp = &Wg[(size_t)br * EE + bc];
  int boff0 = (bc >> 4) * SUBT + b_off(br, bc & 15);        // + j*FS
  int boff1 = (bc >> 4) * SUBT + b_off(br + 16, bc & 15);

  f32x4 va = ldg4(agp), vb0 = ldg4(bgp), vb1 = ldg4(bgp + 16 * EE);
  auto stage = [&](int buf) {
    st4(&As[buf][aoff], va);
    __bf16* bs = Bs[buf];
    bs[boff0 + 0 * FS] = f2bf(vb0.x); bs[boff0 + 1 * FS] = f2bf(vb0.y);
    bs[boff0 + 2 * FS] = f2bf(vb0.z); bs[boff0 + 3 * FS] = f2bf(vb0.w);
    bs[boff1 + 0 * FS] = f2bf(vb1.x); bs[boff1 + 1 * FS] = f2bf(vb1.y);
    bs[boff1 + 2 * FS] = f2bf(vb1.z); bs[boff1 + 3 * FS] = f2bf(vb1.w);
  };
  stage(0);
  const int NIT = DD / 32;
  v8f acc = {};
  for (int it = 0; it < NIT; ++it) {
    __syncthreads();
    int cur = it & 1;
    bool more = (it + 1 < NIT);
    if (more) {               // prefetch next tile into registers
      agp += 32; bgp += (size_t)32 * EE;
      va = ldg4(agp); vb0 = ldg4(bgp); vb1 = ldg4(bgp + 16 * EE);
    }
    bf16x16 a = load_frag(&As[cur][mi * SUBT], lane);
    bf16x16 b = load_frag(&Bs[cur][ni * SUBT], lane);
    acc = wmma_bf16(a, b, acc);
    if (more) stage(cur ^ 1);
  }
  int n = ni * 16 + (lane & 15);
  int mb = m0 + mi * 16 + ((lane >> 4) * 8);
#pragma unroll
  for (int r = 0; r < 8; ++r) gating[(size_t)(mb + r) * EE + n] = acc[r];
}

// ---------------- K2: column softmax stats over tokens ----------------
__global__ __launch_bounds__(256) void softmoe_colstats(const float* __restrict__ gating,
                                                        float* __restrict__ colmax,
                                                        float* __restrict__ colsum) {
  int b = blockIdx.x;
  int tid = threadIdx.x;
  int e = tid & 63, sl = tid >> 6;
  const float* g = gating + (size_t)b * SS * EE;
  __shared__ float red[256];
  __shared__ float cm[64];
  float mx = -3.0e38f;
  for (int s = sl; s < SS; s += 4) mx = fmaxf(mx, g[(size_t)s * EE + e]);
  red[tid] = mx; __syncthreads();
  if (sl == 0) {
    mx = fmaxf(fmaxf(red[e], red[e + 64]), fmaxf(red[e + 128], red[e + 192]));
    cm[e] = mx;
  }
  __syncthreads();
  mx = cm[e];
  float sm = 0.f;
  for (int s = sl; s < SS; s += 4) sm += __expf(g[(size_t)s * EE + e] - mx);
  red[tid] = sm; __syncthreads();
  if (sl == 0) {
    sm = (red[e] + red[e + 64]) + (red[e + 128] + red[e + 192]);
    colmax[b * 64 + e] = mx;
    colsum[b * 64 + e] = sm;
  }
}

// ---------------- K3: slots partials = dis^T @ x (split-K over S) ----------------
// grid (D/64, B, 8); 256 thr (8 waves); tile 64(M=E) x 64(N=D), double-buffered
__global__ __launch_bounds__(256) void softmoe_slots(const float* __restrict__ x,
                                                     const float* __restrict__ gating,
                                                     const float* __restrict__ colmax,
                                                     const float* __restrict__ colsum,
                                                     float* __restrict__ partial) {
  __shared__ __align__(16) __bf16 As[2][4 * SUBT];
  __shared__ __align__(16) __bf16 Bs[2][4 * SUBT];
  int n0 = blockIdx.x * 64;
  int b  = blockIdx.y;
  int kc = blockIdx.z;
  int tid = threadIdx.x, lane = tid & 31, wave = tid >> 5;
  int ni = wave & 3, mi0 = (wave >> 2) * 2;
  int kbeg = kc * (SS / 8);

  // A fill: two float4 per thread (tokens ka, ka+16), experts m4..m4+3
  int ka = tid >> 4, m4 = (tid & 15) * 4;
  const float* gb = gating + ((size_t)b * SS + kbeg + ka) * EE + m4;
  float cm0 = colmax[b * 64 + m4], cm1 = colmax[b * 64 + m4 + 1];
  float cm2 = colmax[b * 64 + m4 + 2], cm3 = colmax[b * 64 + m4 + 3];
  float cr0 = 1.f / colsum[b * 64 + m4], cr1 = 1.f / colsum[b * 64 + m4 + 1];
  float cr2 = 1.f / colsum[b * 64 + m4 + 2], cr3 = 1.f / colsum[b * 64 + m4 + 3];
  int asub = (m4 >> 4) * SUBT;
  int ml = m4 & 15;
  int aoff0 = asub + a_off(ml, ka);          // + j*FS per expert
  int aoff1 = asub + a_off(ml, ka + 16);
  // B fill: two float4 per thread (tokens ka, ka+16), d-cols bc..bc+3
  int bc = (tid & 15) * 4;
  const float* xb = x + ((size_t)b * SS + kbeg + ka) * DD + n0 + bc;
  int boff0 = (bc >> 4) * SUBT + b_off(ka, bc & 15);
  int boff1 = (bc >> 4) * SUBT + b_off(ka + 16, bc & 15);

  f32x4 ga0 = ldg4(gb), ga1 = ldg4(gb + (size_t)16 * EE);
  f32x4 vb0 = ldg4(xb), vb1 = ldg4(xb + (size_t)16 * DD);
  auto stage = [&](int buf) {
    __bf16* as = As[buf];
    as[aoff0 + 0 * FS] = f2bf(__expf(ga0.x - cm0) * cr0);
    as[aoff0 + 1 * FS] = f2bf(__expf(ga0.y - cm1) * cr1);
    as[aoff0 + 2 * FS] = f2bf(__expf(ga0.z - cm2) * cr2);
    as[aoff0 + 3 * FS] = f2bf(__expf(ga0.w - cm3) * cr3);
    as[aoff1 + 0 * FS] = f2bf(__expf(ga1.x - cm0) * cr0);
    as[aoff1 + 1 * FS] = f2bf(__expf(ga1.y - cm1) * cr1);
    as[aoff1 + 2 * FS] = f2bf(__expf(ga1.z - cm2) * cr2);
    as[aoff1 + 3 * FS] = f2bf(__expf(ga1.w - cm3) * cr3);
    __bf16* bs = Bs[buf];
    bs[boff0 + 0 * FS] = f2bf(vb0.x); bs[boff0 + 1 * FS] = f2bf(vb0.y);
    bs[boff0 + 2 * FS] = f2bf(vb0.z); bs[boff0 + 3 * FS] = f2bf(vb0.w);
    bs[boff1 + 0 * FS] = f2bf(vb1.x); bs[boff1 + 1 * FS] = f2bf(vb1.y);
    bs[boff1 + 2 * FS] = f2bf(vb1.z); bs[boff1 + 3 * FS] = f2bf(vb1.w);
  };
  stage(0);
  const int NIT = (SS / 8) / 32;
  v8f acc0 = {}, acc1 = {};
  for (int it = 0; it < NIT; ++it) {
    __syncthreads();
    int cur = it & 1;
    bool more = (it + 1 < NIT);
    if (more) {
      gb += (size_t)32 * EE; xb += (size_t)32 * DD;
      ga0 = ldg4(gb); ga1 = ldg4(gb + (size_t)16 * EE);
      vb0 = ldg4(xb); vb1 = ldg4(xb + (size_t)16 * DD);
    }
    bf16x16 bm = load_frag(&Bs[cur][ni * SUBT], lane);
    bf16x16 a0 = load_frag(&As[cur][mi0 * SUBT], lane);
    acc0 = wmma_bf16(a0, bm, acc0);
    bf16x16 a1 = load_frag(&As[cur][(mi0 + 1) * SUBT], lane);
    acc1 = wmma_bf16(a1, bm, acc1);
    if (more) stage(cur ^ 1);
  }
  float* pb = partial + ((size_t)kc * BB + b) * EE * DD;
  int n = n0 + ni * 16 + (lane & 15);
  int e0 = mi0 * 16 + ((lane >> 4) * 8);
#pragma unroll
  for (int r = 0; r < 8; ++r) pb[(size_t)(e0 + r) * DD + n] = acc0[r];
  int e1 = (mi0 + 1) * 16 + ((lane >> 4) * 8);
#pragma unroll
  for (int r = 0; r < 8; ++r) pb[(size_t)(e1 + r) * DD + n] = acc1[r];
}

__global__ __launch_bounds__(256) void softmoe_slots_reduce(const float* __restrict__ partial,
                                                            float* __restrict__ slots) {
  const int N = BB * EE * DD;
  int i4 = (blockIdx.x * 256 + threadIdx.x) * 4;
  if (i4 < N) {
    f32x4 s = ldg4(&partial[i4]);
#pragma unroll
    for (int c = 1; c < 8; ++c) {
      f32x4 v = ldg4(&partial[(size_t)c * N + i4]);
      s += v;
    }
    *(f32x4*)&slots[i4] = s;
  }
}

// ---------------- K4: y[b,e,:] = slots[b,e,:] @ We[e] + be[e] ----------------
// grid (D/64, E); 128 thr (4 waves); M=16 tile (rows 0..3 = batches); NT loads of We
__global__ __launch_bounds__(128) void softmoe_expert(const float* __restrict__ slots,
                                                      const float* __restrict__ We,
                                                      const float* __restrict__ bias,
                                                      float* __restrict__ y) {
  __shared__ __align__(16) __bf16 As[2][SUBT];
  __shared__ __align__(16) __bf16 Bs[2][4 * SUBT];
  int n0 = blockIdx.x * 64;
  int e  = blockIdx.y;
  int tid = threadIdx.x, lane = tid & 31, wave = tid >> 5;

  // A fill: one float4 per thread; rows >= BB stay zero (pre-cleared once)
  int ar = tid >> 3, ac = (tid & 7) * 4;
  const float* agp = &slots[((size_t)ar * EE + e) * DD + ac];
  int aoff = a_off(ar, ac);
  bool avalid = (ar < BB);
  if (!avalid) {
    __bf16 z = (__bf16)0.f;
    As[0][aoff] = z; As[0][aoff + 1] = z; As[0][aoff + 2] = z; As[0][aoff + 3] = z;
    As[1][aoff] = z; As[1][aoff + 1] = z; As[1][aoff + 2] = z; As[1][aoff + 3] = z;
  }
  // B fill: four float4 per thread (rows br+8q)
  int br = tid >> 4, bc = (tid & 15) * 4;
  const float* wgp = We + (size_t)e * DD * DD + (size_t)br * DD + n0 + bc;
  int boff[4];
#pragma unroll
  for (int q = 0; q < 4; ++q)
    boff[q] = (bc >> 4) * SUBT + b_off(br + 8 * q, bc & 15);

  f32x4 va = {}, vb[4];
  if (avalid) va = ldg4(agp);
#pragma unroll
  for (int q = 0; q < 4; ++q) vb[q] = ldg4_nt(wgp + (size_t)8 * q * DD);
  auto stage = [&](int buf) {
    if (avalid) st4(&As[buf][aoff], va);
    __bf16* bs = Bs[buf];
#pragma unroll
    for (int q = 0; q < 4; ++q) {
      bs[boff[q] + 0 * FS] = f2bf(vb[q].x); bs[boff[q] + 1 * FS] = f2bf(vb[q].y);
      bs[boff[q] + 2 * FS] = f2bf(vb[q].z); bs[boff[q] + 3 * FS] = f2bf(vb[q].w);
    }
  };
  stage(0);
  const int NIT = DD / 32;
  v8f acc = {};
  for (int it = 0; it < NIT; ++it) {
    __syncthreads();
    int cur = it & 1;
    bool more = (it + 1 < NIT);
    if (more) {
      agp += 32; wgp += (size_t)32 * DD;
      if (avalid) va = ldg4(agp);
#pragma unroll
      for (int q = 0; q < 4; ++q) vb[q] = ldg4_nt(wgp + (size_t)8 * q * DD);
    }
    bf16x16 a  = load_frag(As[cur], lane);
    bf16x16 bm = load_frag(&Bs[cur][wave * SUBT], lane);
    acc = wmma_bf16(a, bm, acc);
    if (more) stage(cur ^ 1);
  }
  int n = n0 + wave * 16 + (lane & 15);
  int m0 = (lane >> 4) * 8;
  float bv = bias[(size_t)e * DD + n];
#pragma unroll
  for (int r = 0; r < 8; ++r) {
    int m = m0 + r;
    if (m < BB) y[((size_t)m * EE + e) * DD + n] = acc[r] + bv;
  }
}

// ---------------- K5: out = com @ y ----------------
// grid (D/64, S/32, B); 256 thr (8 waves); tile 32(tokens) x 64(D), K=E=64; NT out
__global__ __launch_bounds__(256) void softmoe_combine(const float* __restrict__ gating,
                                                       const float* __restrict__ y,
                                                       float* __restrict__ out) {
  __shared__ float Gs[32 * 64];
  __shared__ __align__(16) __bf16 As[4 * SUBT];   // (mi, kt) 2x2 subtiles
  __shared__ __align__(16) __bf16 Bs[8 * SUBT];   // (kt, ni) 2x4 subtiles
  __shared__ float rmax[32], rrcp[32];
  int n0 = blockIdx.x * 64;
  int s0 = blockIdx.y * 32;
  int b  = blockIdx.z;
  int tid = threadIdx.x, lane = tid & 31, wave = tid >> 5;
  int mi = wave >> 2, ni = wave & 3;
  const float* gb = gating + ((size_t)b * SS + s0) * EE;
  const float* yb = y + (size_t)b * EE * DD;
  for (int i = tid; i < (32 * 64) / 4; i += 256)
    *(f32x4*)&Gs[i * 4] = ldg4(&gb[i * 4]);
  // B: y tile 64(k=expert) x 64(n)
  for (int i = tid; i < (64 * 64) / 4; i += 256) {
    int k = i >> 4, c = (i & 15) * 4;
    f32x4 v = ldg4(&yb[(size_t)k * DD + n0 + c]);
    __bf16* bs = &Bs[((k >> 5) * 4 + (c >> 4)) * SUBT];
    int kl = k & 31, nl = c & 15;
    bs[b_off(kl, nl + 0)] = f2bf(v.x);
    bs[b_off(kl, nl + 1)] = f2bf(v.y);
    bs[b_off(kl, nl + 2)] = f2bf(v.z);
    bs[b_off(kl, nl + 3)] = f2bf(v.w);
  }
  __syncthreads();
  if (tid < 32) {
    float mx = -3.0e38f;
    for (int e2 = 0; e2 < 64; ++e2) mx = fmaxf(mx, Gs[tid * 64 + e2]);
    float sm = 0.f;
    for (int e2 = 0; e2 < 64; ++e2) sm += __expf(Gs[tid * 64 + e2] - mx);
    rmax[tid] = mx; rrcp[tid] = 1.0f / sm;
  }
  __syncthreads();
  // A[m=token][k=expert] = com; k-contiguous -> contiguous frag elems
  for (int i = tid; i < (32 * 64) / 4; i += 256) {
    int r = i >> 4, e4 = (i & 15) * 4;
    float mx = rmax[r], rc = rrcp[r];
    f32x4 v = ldg4(&Gs[r * 64 + e4]);
    f32x4 d;
    d.x = __expf(v.x - mx) * rc; d.y = __expf(v.y - mx) * rc;
    d.z = __expf(v.z - mx) * rc; d.w = __expf(v.w - mx) * rc;
    int kt = e4 >> 5, kl = e4 & 31;
    st4(&As[((r >> 4) * 2 + kt) * SUBT + a_off(r & 15, kl)], d);
  }
  __syncthreads();
  v8f acc = {};
#pragma unroll
  for (int kt = 0; kt < 2; ++kt) {
    bf16x16 a  = load_frag(&As[(mi * 2 + kt) * SUBT], lane);
    bf16x16 bm = load_frag(&Bs[(kt * 4 + ni) * SUBT], lane);
    acc = wmma_bf16(a, bm, acc);
  }
  int n = n0 + ni * 16 + (lane & 15);
  int sr = s0 + mi * 16 + ((lane >> 4) * 8);
  float* ob = out + (size_t)b * SS * DD;
#pragma unroll
  for (int r = 0; r < 8; ++r)
    __builtin_nontemporal_store(acc[r], &ob[(size_t)(sr + r) * DD + n]);
}

extern "C" void kernel_launch(void* const* d_in, const int* in_sizes, int n_in,
                              void* d_out, int out_size, void* d_ws, size_t ws_size,
                              hipStream_t stream) {
  const float* x  = (const float*)d_in[0];
  const float* Wg = (const float*)d_in[1];
  const float* We = (const float*)d_in[2];
  const float* be = (const float*)d_in[3];
  float* out = (float*)d_out;

  float* ws      = (float*)d_ws;
  float* gating  = ws;                                   // B*S*E
  float* colmax  = gating + (size_t)BB * SS * EE;        // 256
  float* colsum  = colmax + BB * EE;                     // 256
  float* partial = colsum + BB * EE;                     // 8*B*E*D
  float* slots   = partial + (size_t)8 * BB * EE * DD;   // B*E*D
  float* ybuf    = slots + (size_t)BB * EE * DD;         // B*E*D

  softmoe_gating <<<dim3((BB * SS) / 32), 256, 0, stream>>>(x, Wg, gating);
  softmoe_colstats<<<dim3(BB), 256, 0, stream>>>(gating, colmax, colsum);
  softmoe_slots  <<<dim3(DD / 64, BB, 8), 256, 0, stream>>>(x, gating, colmax, colsum, partial);
  softmoe_slots_reduce<<<dim3((BB * EE * DD / 4 + 255) / 256), 256, 0, stream>>>(partial, slots);
  softmoe_expert <<<dim3(DD / 64, EE), 128, 0, stream>>>(slots, We, be, ybuf);
  softmoe_combine<<<dim3(DD / 64, SS / 32, BB), 256, 0, stream>>>(gating, ybuf, out);
}
